// CrossModalAttention_79697413144803
// MI455X (gfx1250) — matrix operand order repaired and checked by reading, not compiled, not run on hardware
//
#include <hip/hip_runtime.h>
#include <stdint.h>

// ---------------------------------------------------------------------------
// Problem sizes (fixed by the reference)
// ---------------------------------------------------------------------------
static constexpr int BB  = 32;     // batch
static constexpr int NN  = 1024;   // H*W tokens
static constexpr int CCN = 512;    // C_CNN
static constexpr int CEF = 1280;   // C_EFF
static constexpr int HD  = 512;    // HID

typedef __bf16 bf16;
typedef __attribute__((ext_vector_type(16))) __bf16 v16bf;
typedef __attribute__((ext_vector_type(8)))  float  v8f;

// ---------------------------------------------------------------------------
// Helpers
// ---------------------------------------------------------------------------
__device__ __forceinline__ bf16 to_bf16(float f) {
    union { float f; uint32_t u; } x; x.f = f;
    uint32_t r = x.u + 0x7FFFu + ((x.u >> 16) & 1u);   // round-to-nearest-even
    union { uint16_t u; bf16 b; } y; y.u = (uint16_t)(r >> 16);
    return y.b;
}

// A fragment: 16x32 (MxK) bf16 from row-major [M][K] source.
// ISA layout: lanes 0-15 hold row M=lane, K = {k0..k0+7, k0+16..k0+23};
//             lanes 16-31 hold row M=lane-16, K = {k0+8..k0+15, k0+24..k0+31}.
__device__ __forceinline__ v16bf load_frag_A(const bf16* __restrict__ base,
                                             int pitch, int m0, int k0, int lane) {
    const bf16* p = base + (size_t)(m0 + (lane & 15)) * pitch + k0 + (lane >> 4) * 8;
    union { uint4 u[2]; v16bf v; } r;
    r.u[0] = *(const uint4*)(p);
    r.u[1] = *(const uint4*)(p + 16);
    return r.v;
}

// B fragment: 32x16 (KxN) bf16 taken from a TRANSPOSED source stored [N][K].
// ISA layout: lanes 0-15 hold col N=lane, K = {k0..k0+7, k0+16..k0+23};
//             lanes 16-31 hold col N=lane-16, K = {k0+8..k0+15, k0+24..k0+31}.
__device__ __forceinline__ v16bf load_frag_B(const bf16* __restrict__ baseT,
                                             int pitch, int n0, int k0, int lane) {
    const bf16* p = baseT + (size_t)(n0 + (lane & 15)) * pitch + k0 + (lane >> 4) * 8;
    union { uint4 u[2]; v16bf v; } r;
    r.u[0] = *(const uint4*)(p);
    r.u[1] = *(const uint4*)(p + 16);
    return r.v;
}

__device__ __forceinline__ v8f wmma_bf16(v16bf a, v16bf b, v8f c) {
    return __builtin_amdgcn_wmma_f32_16x16x32_bf16(false, a, false, b,
                                                   (short)0, c, false, false);
}

// ---------------------------------------------------------------------------
// Kernel: f32 -> bf16 weight conversion
// ---------------------------------------------------------------------------
__global__ void cvt_f32_bf16(const float* __restrict__ src, bf16* __restrict__ dst, int n) {
    int i = blockIdx.x * blockDim.x + threadIdx.x;
    if (i < n) dst[i] = to_bf16(src[i]);
}

// ---------------------------------------------------------------------------
// Kernel: transpose+convert  x[b][C][N] f32  ->  xT[b][N][C] bf16
// grid (C/32, N/32, B), block (32, 8)
// ---------------------------------------------------------------------------
__global__ void transpose_bf16(const float* __restrict__ x, bf16* __restrict__ xT,
                               int Cdim, int Ndim) {
    __shared__ bf16 tile[32][33];
    const int b  = blockIdx.z;
    const float* xb  = x  + (size_t)b * Cdim * Ndim;
    bf16*        xTb = xT + (size_t)b * Cdim * Ndim;
    const int c0 = blockIdx.x * 32, n0 = blockIdx.y * 32;
    const int tx = threadIdx.x, ty = threadIdx.y;
#pragma unroll
    for (int i = 0; i < 4; i++) {
        int c = ty + 8 * i;
        tile[c][tx] = to_bf16(xb[(size_t)(c0 + c) * Ndim + n0 + tx]);
    }
    __syncthreads();
#pragma unroll
    for (int i = 0; i < 4; i++) {
        int n = ty + 8 * i;
        xTb[(size_t)(n0 + n) * Cdim + c0 + tx] = tile[tx][n];
    }
}

// ---------------------------------------------------------------------------
// Kernel: batched WMMA GEMM   C[M][Nc] = A[M][K] * Bt[Nc][K]^T + bias[M]
// A (weights) shared across batch; Bt batched with stride Nc*K.
// Workgroup = 128x128 tile, 8 waves, each wave owns a 32x64 strip
// (2 m-tiles x 4 n-tiles = 8 independent WMMA accumulator chains; per K-step
//  2 A-frags + 4 B-frags feed 8 WMMAs -> 1.5 b128 loads per WMMA).
// Outputs (any may be null):
//   outT : bf16 [Nc][M]  (transposed -> B-side-ready for the next GEMM)
//   outN : bf16 [M][Nc]
//   outF : f32  [M][Nc]
// grid (M/128, Nc/128, B), block 256
// ---------------------------------------------------------------------------
__global__ __launch_bounds__(256)
void gemm_wmma(const bf16* __restrict__ A, const bf16* __restrict__ Bt,
               const float* __restrict__ bias,
               bf16* __restrict__ outT, bf16* __restrict__ outN, float* __restrict__ outF,
               int M, int Nc, int K) {
    const int b    = blockIdx.z;
    const bf16* BtB = Bt + (size_t)b * Nc * K;
    const int lane = threadIdx.x & 31;
    const int wave = threadIdx.x >> 5;          // 0..7
    const int m_base = blockIdx.x * 128 + (wave >> 1) * 32;  // 4 m-strips
    const int n_base = blockIdx.y * 128 + (wave & 1) * 64;   // 2 n-strips

    v8f acc[2][4];
#pragma unroll
    for (int i = 0; i < 2; i++)
#pragma unroll
        for (int j = 0; j < 4; j++)
            acc[i][j] = (v8f){0.f, 0.f, 0.f, 0.f, 0.f, 0.f, 0.f, 0.f};

    for (int k0 = 0; k0 < K; k0 += 32) {
        v16bf a0 = load_frag_A(A, K, m_base,      k0, lane);
        v16bf a1 = load_frag_A(A, K, m_base + 16, k0, lane);
        v16bf b0 = load_frag_B(BtB, K, n_base,      k0, lane);
        v16bf b1 = load_frag_B(BtB, K, n_base + 16, k0, lane);
        v16bf b2 = load_frag_B(BtB, K, n_base + 32, k0, lane);
        v16bf b3 = load_frag_B(BtB, K, n_base + 48, k0, lane);
        acc[0][0] = wmma_bf16(a0, b0, acc[0][0]);
        acc[0][1] = wmma_bf16(a0, b1, acc[0][1]);
        acc[0][2] = wmma_bf16(a0, b2, acc[0][2]);
        acc[0][3] = wmma_bf16(a0, b3, acc[0][3]);
        acc[1][0] = wmma_bf16(a1, b0, acc[1][0]);
        acc[1][1] = wmma_bf16(a1, b1, acc[1][1]);
        acc[1][2] = wmma_bf16(a1, b2, acc[1][2]);
        acc[1][3] = wmma_bf16(a1, b3, acc[1][3]);
    }

    // D layout: lane holds col n = n0+(lane&15); rows m = m0 + 8*(lane>>4) + r
    const int half8 = 8 * (lane >> 4);
#pragma unroll
    for (int i = 0; i < 2; i++) {
        const int m0 = m_base + 16 * i;
#pragma unroll
        for (int j = 0; j < 4; j++) {
            const int n0 = n_base + 16 * j;
            const int n  = n0 + (lane & 15);
            float vals[8];
#pragma unroll
            for (int r = 0; r < 8; r++) vals[r] = acc[i][j][r] + bias[m0 + half8 + r];

            if (outT) {  // contiguous 16B store per lane
                union { bf16 h[8]; uint4 u; } pk;
#pragma unroll
                for (int r = 0; r < 8; r++) pk.h[r] = to_bf16(vals[r]);
                bf16* o = outT + (size_t)b * Nc * M + (size_t)n * M + m0 + half8;
                *(uint4*)o = pk.u;
            }
            if (outN) {
                bf16* o = outN + (size_t)b * M * Nc;
#pragma unroll
                for (int r = 0; r < 8; r++)
                    o[(size_t)(m0 + half8 + r) * Nc + n] = to_bf16(vals[r]);
            }
            if (outF) {
                float* o = outF + (size_t)b * M * Nc;
#pragma unroll
                for (int r = 0; r < 8; r++)
                    o[(size_t)(m0 + half8 + r) * Nc + n] = vals[r];
            }
        }
    }
}

// ---------------------------------------------------------------------------
// Kernel: fused attention for 16 query tokens per workgroup.
//   S = q kT  (WMMA, f32 -> LDS), row softmax, P bf16 -> LDS,
//   O = P V   (WMMA), out = gamma*O^T + cnn_proj (f32).
// qT,kT: [B][N][HD] bf16 ; v: [B][HD][N] bf16 ; cnn_proj/out: [B][HD][N] f32
// grid (N/16, B), block 256, dyn LDS = 16*1024*4 + 16*1024*2 = 96 KB
// ---------------------------------------------------------------------------
__global__ __launch_bounds__(256)
void attn_kernel(const bf16* __restrict__ qT, const bf16* __restrict__ kT,
                 const bf16* __restrict__ v,  const float* __restrict__ cnn_proj,
                 const float* __restrict__ gamma, float* __restrict__ out) {
    extern __shared__ char smem[];
    float* S = (float*)smem;                              // [16][1024]
    bf16*  P = (bf16*)(smem + 16 * 1024 * sizeof(float)); // [16][1024]

    const int b  = blockIdx.y;
    const int q0 = blockIdx.x * 16;
    const bf16*  qTb   = qT + (size_t)b * NN * HD;
    const bf16*  kTb   = kT + (size_t)b * NN * HD;
    const bf16*  vb    = v  + (size_t)b * HD * NN;
    const float* projb = cnn_proj + (size_t)b * HD * NN;
    float*       outb  = out      + (size_t)b * HD * NN;

    const int lane = threadIdx.x & 31;
    const int wave = threadIdx.x >> 5;
    const int nl   = lane & 15;
    const int hf   = lane >> 4;

    // ---- Phase 1: S[16][1024] = q(16xHD) * k(HDx1024)
    // Each wave: 8 n-tiles as 2 passes of 4 concurrent accumulators,
    // A-fragment shared across all 4 WMMAs of a K-step.
#pragma unroll
    for (int pass = 0; pass < 2; pass++) {
        const int ntb = wave * 8 + pass * 4;
        v8f acc[4];
#pragma unroll
        for (int j = 0; j < 4; j++)
            acc[j] = (v8f){0.f, 0.f, 0.f, 0.f, 0.f, 0.f, 0.f, 0.f};

        for (int k0 = 0; k0 < HD; k0 += 32) {
            v16bf a  = load_frag_A(qTb, HD, q0, k0, lane);
            v16bf b0 = load_frag_B(kTb, HD, (ntb + 0) * 16, k0, lane);
            v16bf b1 = load_frag_B(kTb, HD, (ntb + 1) * 16, k0, lane);
            v16bf b2 = load_frag_B(kTb, HD, (ntb + 2) * 16, k0, lane);
            v16bf b3 = load_frag_B(kTb, HD, (ntb + 3) * 16, k0, lane);
            acc[0] = wmma_bf16(a, b0, acc[0]);
            acc[1] = wmma_bf16(a, b1, acc[1]);
            acc[2] = wmma_bf16(a, b2, acc[2]);
            acc[3] = wmma_bf16(a, b3, acc[3]);
        }
#pragma unroll
        for (int j = 0; j < 4; j++)
#pragma unroll
            for (int r = 0; r < 8; r++)
                S[(r + 8 * hf) * 1024 + (ntb + j) * 16 + nl] = acc[j][r];
    }
    __syncthreads();

    // ---- Phase 2: softmax over each row (16 threads per row, shfl_xor tree)
    {
        const int row = threadIdx.x >> 4;   // 0..15
        const int cl  = threadIdx.x & 15;
        float mx = -3.0e38f;
#pragma unroll 8
        for (int i = 0; i < 64; i++) mx = fmaxf(mx, S[row * 1024 + cl + 16 * i]);
        for (int d = 1; d < 16; d <<= 1) mx = fmaxf(mx, __shfl_xor(mx, d, 32));
        float sum = 0.f;
#pragma unroll 8
        for (int i = 0; i < 64; i++) {
            int idx = row * 1024 + cl + 16 * i;
            float e = __expf(S[idx] - mx);
            S[idx] = e; sum += e;
        }
        for (int d = 1; d < 16; d <<= 1) sum += __shfl_xor(sum, d, 32);
        const float inv = 1.0f / sum;
#pragma unroll 8
        for (int i = 0; i < 64; i++) {
            int idx = row * 1024 + cl + 16 * i;
            P[idx] = to_bf16(S[idx] * inv);
        }
    }
    __syncthreads();

    // ---- Phase 3: O = P(16x1024) * V(1024xHD)
    // 4 concurrent c-tile accumulators per wave; LDS A-fragment shared by all 4.
    const float g = gamma[0];
    v8f acc[4];
#pragma unroll
    for (int j = 0; j < 4; j++)
        acc[j] = (v8f){0.f, 0.f, 0.f, 0.f, 0.f, 0.f, 0.f, 0.f};

    for (int k0 = 0; k0 < NN; k0 += 32) {
        // A fragment straight from LDS P (row-major [16][1024])
        const bf16* pa = P + nl * 1024 + k0 + hf * 8;
        union { uint4 u[2]; v16bf v; } ua;
        ua.u[0] = *(const uint4*)(pa);
        ua.u[1] = *(const uint4*)(pa + 16);
        v16bf b0 = load_frag_B(vb, NN, (wave +  0) * 16, k0, lane);
        v16bf b1 = load_frag_B(vb, NN, (wave +  8) * 16, k0, lane);
        v16bf b2 = load_frag_B(vb, NN, (wave + 16) * 16, k0, lane);
        v16bf b3 = load_frag_B(vb, NN, (wave + 24) * 16, k0, lane);
        acc[0] = wmma_bf16(ua.v, b0, acc[0]);
        acc[1] = wmma_bf16(ua.v, b1, acc[1]);
        acc[2] = wmma_bf16(ua.v, b2, acc[2]);
        acc[3] = wmma_bf16(ua.v, b3, acc[3]);
    }

#pragma unroll
    for (int j = 0; j < 4; j++) {
        // D: lane -> channel c = ct*16 + nl ; rows = tokens q0 + 8*hf + r (contig)
        const int ct  = wave + 8 * j;
        const int c   = ct * 16 + nl;
        const int tkn = q0 + 8 * hf;
        const size_t base = (size_t)c * NN + tkn;
        float4 p0 = *(const float4*)(projb + base);
        float4 p1 = *(const float4*)(projb + base + 4);
        float4 o0, o1;
        o0.x = g * acc[j][0] + p0.x;  o0.y = g * acc[j][1] + p0.y;
        o0.z = g * acc[j][2] + p0.z;  o0.w = g * acc[j][3] + p0.w;
        o1.x = g * acc[j][4] + p1.x;  o1.y = g * acc[j][5] + p1.y;
        o1.z = g * acc[j][6] + p1.z;  o1.w = g * acc[j][7] + p1.w;
        *(float4*)(outb + base)     = o0;
        *(float4*)(outb + base + 4) = o1;
    }
}

// ---------------------------------------------------------------------------
// Host launcher
// ---------------------------------------------------------------------------
extern "C" void kernel_launch(void* const* d_in, const int* in_sizes, int n_in,
                              void* d_out, int out_size, void* d_ws, size_t ws_size,
                              hipStream_t stream) {
    (void)in_sizes; (void)n_in; (void)out_size; (void)ws_size;

    const float* cnn_x  = (const float*)d_in[0];
    const float* eff_x  = (const float*)d_in[1];
    const float* W_cnn  = (const float*)d_in[2];
    const float* b_cnn  = (const float*)d_in[3];
    const float* W_eff  = (const float*)d_in[4];
    const float* b_eff  = (const float*)d_in[5];
    const float* W_q    = (const float*)d_in[6];
    const float* b_q    = (const float*)d_in[7];
    const float* W_k    = (const float*)d_in[8];
    const float* b_k    = (const float*)d_in[9];
    const float* W_v    = (const float*)d_in[10];
    const float* b_v    = (const float*)d_in[11];
    const float* gamma  = (const float*)d_in[12];
    float* out = (float*)d_out;

    // ---- workspace carve (256B aligned)
    char*  ws  = (char*)d_ws;
    size_t off = 0;
    auto carve = [&](size_t bytes) -> char* {
        char* p = ws + off;
        off += (bytes + 255) & ~(size_t)255;
        return p;
    };
    bf16*  cnnT   = (bf16*)carve((size_t)BB * NN * CCN * 2); // inputs transposed
    bf16*  effT   = (bf16*)carve((size_t)BB * NN * CEF * 2);
    bf16*  Wc     = (bf16*)carve((size_t)HD * CCN * 2);      // bf16 weights
    bf16*  We     = (bf16*)carve((size_t)HD * CEF * 2);
    bf16*  Wq     = (bf16*)carve((size_t)HD * HD * 2);
    bf16*  Wk     = (bf16*)carve((size_t)HD * HD * 2);
    bf16*  Wv     = (bf16*)carve((size_t)HD * HD * 2);
    bf16*  cnnPT  = (bf16*)carve((size_t)BB * NN * HD * 2);  // cnn_proj^T bf16
    float* cnnPF  = (float*)carve((size_t)BB * HD * NN * 4); // cnn_proj f32 (residual)
    bf16*  effPT  = (bf16*)carve((size_t)BB * NN * HD * 2);  // eff_proj^T bf16
    bf16*  qTbuf  = (bf16*)carve((size_t)BB * NN * HD * 2);  // q^T  [N][C]
    bf16*  kTbuf  = (bf16*)carve((size_t)BB * NN * HD * 2);  // k^T  [N][C]
    bf16*  vbuf   = (bf16*)carve((size_t)BB * HD * NN * 2);  // v    [C][N]

    // ---- 1. weight conversion f32->bf16
    auto cvt = [&](const float* s, bf16* d, int n) {
        cvt_f32_bf16<<<(n + 255) / 256, 256, 0, stream>>>(s, d, n);
    };
    cvt(W_cnn, Wc, HD * CCN);
    cvt(W_eff, We, HD * CEF);
    cvt(W_q,   Wq, HD * HD);
    cvt(W_k,   Wk, HD * HD);
    cvt(W_v,   Wv, HD * HD);

    // ---- 2. transpose inputs to [N][C] bf16
    transpose_bf16<<<dim3(CCN / 32, NN / 32, BB), dim3(32, 8), 0, stream>>>(cnn_x, cnnT, CCN, NN);
    transpose_bf16<<<dim3(CEF / 32, NN / 32, BB), dim3(32, 8), 0, stream>>>(eff_x, effT, CEF, NN);

    const dim3 gblk(256);
    const dim3 ggrd(HD / 128, NN / 128, BB);

    // ---- 3. projections (all WMMA)
    // cnn_proj: transposed bf16 (feeds q GEMM) + f32 (residual)
    gemm_wmma<<<ggrd, gblk, 0, stream>>>(Wc, cnnT, b_cnn, cnnPT, (bf16*)nullptr, cnnPF,
                                         HD, NN, CCN);
    // eff_proj: transposed bf16 (feeds k/v GEMMs)
    gemm_wmma<<<ggrd, gblk, 0, stream>>>(We, effT, b_eff, effPT, (bf16*)nullptr, (float*)nullptr,
                                         HD, NN, CEF);
    // q^T, k^T: [N][C] (A-side / B-side of attention score GEMM)
    gemm_wmma<<<ggrd, gblk, 0, stream>>>(Wq, cnnPT, b_q, qTbuf, (bf16*)nullptr, (float*)nullptr,
                                         HD, NN, HD);
    gemm_wmma<<<ggrd, gblk, 0, stream>>>(Wk, effPT, b_k, kTbuf, (bf16*)nullptr, (float*)nullptr,
                                         HD, NN, HD);
    // v: natural [C][N] (B-side of P@V)
    gemm_wmma<<<ggrd, gblk, 0, stream>>>(Wv, effPT, b_v, (bf16*)nullptr, vbuf, (float*)nullptr,
                                         HD, NN, HD);

    // ---- 4. fused attention + residual epilogue
    const size_t smem = (size_t)16 * 1024 * sizeof(float) + (size_t)16 * 1024 * sizeof(bf16);
    attn_kernel<<<dim3(NN / 16, BB), dim3(256), smem, stream>>>(qTbuf, kTbuf, vbuf,
                                                                cnnPF, gamma, out);
}
